// GRU_43568148251487
// MI455X (gfx1250) — matrix operand order repaired and checked by reading, not compiled, not run on hardware
//
#include <hip/hip_runtime.h>
#include <hip/hip_bf16.h>

// ---------------------------------------------------------------------------
// Problem constants (from the reference)
// ---------------------------------------------------------------------------
#define BB      64      // batch
#define TT      512     // time steps
#define D_IN    64
#define D_MODEL 256
#define HH      512     // hidden
#define G3H     1536    // 3*H
#define D_OUT   64
#define ROWS    (BB * TT)   // 32768

typedef __attribute__((ext_vector_type(16))) __bf16 v16bf;
typedef __attribute__((ext_vector_type(8)))  float  v8f;

// ---------------------------------------------------------------------------
// Utility kernels
// ---------------------------------------------------------------------------
__global__ __launch_bounds__(256)
void k_f32_to_bf16(const float* __restrict__ src, __bf16* __restrict__ dst, int n) {
    int i = blockIdx.x * 256 + threadIdx.x;
    if (i < n) dst[i] = (__bf16)src[i];
}

__global__ __launch_bounds__(256)
void k_zero_u32(unsigned int* __restrict__ p, int n) {
    int i = blockIdx.x * 256 + threadIdx.x;
    if (i < n) p[i] = 0u;
}

// ---------------------------------------------------------------------------
// Input projection: xp[row, m] = x[row,:]·Wx[m,:] + bx[m] + t[row]*Wt[m] + bt[m]
// row = b*T + t ; stored directly as bf16 (WMMA A operand for gi0 GEMM).
// ---------------------------------------------------------------------------
__global__ __launch_bounds__(256)
void k_input_proj(const float* __restrict__ x, const float* __restrict__ tin,
                  const float* __restrict__ Wx, const float* __restrict__ bx,
                  const float* __restrict__ Wt, const float* __restrict__ bt,
                  __bf16* __restrict__ xpb) {
    int idx = blockIdx.x * 256 + threadIdx.x;       // over ROWS * D_MODEL
    int m   = idx & (D_MODEL - 1);
    int row = idx >> 8;                             // D_MODEL == 256
    const float* xr = x  + (size_t)row * D_IN;
    const float* wr = Wx + (size_t)m   * D_IN;
    float acc = bx[m] + bt[m] + tin[row] * Wt[m];   // D_TIME == 1
#pragma unroll
    for (int k = 0; k < D_IN; ++k) acc += xr[k] * wr[k];
    xpb[(size_t)row * D_MODEL + m] = (__bf16)acc;
}

// ---------------------------------------------------------------------------
// Generic bf16 WMMA GEMM:  C[M,N] = A[M,K] * W[N,K]^T + bias[N]   (f32 out)
// Block tile 64(M) x 64(N); 8 waves; each wave a 16x32 tile (2 accumulators
// sharing one A fragment -> 2 WMMA per A load).
// Operand k order used here (contiguous, same for A and B):
//   A: lane L -> row m0+L%16, K = k0 + (L<16?0:16) + {0..15} (contig in A row)
//   B: lane L -> col n0+L%16, K = k0 + (L<16?0:16) + {0..15} (contig in W row)
// ---------------------------------------------------------------------------
__global__ __launch_bounds__(256)
void k_gemm_bf16(const __bf16* __restrict__ A, const __bf16* __restrict__ W,
                 const float* __restrict__ bias, float* __restrict__ C,
                 int K, int N) {
    const int tid  = threadIdx.x;
    const int lane = tid & 31;
    const int wave = tid >> 5;
    const int lc   = lane & 15;
    const int hi   = lane >> 4;
    const int m0   = blockIdx.x * 64 + (wave >> 1) * 16;
    const int n0   = blockIdx.y * 64 + (wave & 1) * 32;

    const __bf16* arow = A + (size_t)(m0 + lc) * K + hi * 16;
    const __bf16* w0   = W + (size_t)(n0 + lc)      * K + hi * 16;
    const __bf16* w1   = W + (size_t)(n0 + 16 + lc) * K + hi * 16;

    v8f acc0 = {}, acc1 = {};
    for (int k0 = 0; k0 < K; k0 += 32) {
        v16bf a, b0, b1;
        *(uint4*)&a         = *(const uint4*)(arow + k0);
        *((uint4*)&a + 1)   = *(const uint4*)(arow + k0 + 8);
        *(uint4*)&b0        = *(const uint4*)(w0 + k0);
        *((uint4*)&b0 + 1)  = *(const uint4*)(w0 + k0 + 8);
        *(uint4*)&b1        = *(const uint4*)(w1 + k0);
        *((uint4*)&b1 + 1)  = *(const uint4*)(w1 + k0 + 8);
        if (k0 + 64 < K) {
            __builtin_prefetch(arow + k0 + 64, 0, 1);
            __builtin_prefetch(w0 + k0 + 64, 0, 1);
            __builtin_prefetch(w1 + k0 + 64, 0, 1);
        }
        acc0 = __builtin_amdgcn_wmma_f32_16x16x32_bf16(
                   false, a, false, b0, (short)0, acc0, false, false);
        acc1 = __builtin_amdgcn_wmma_f32_16x16x32_bf16(
                   false, a, false, b1, (short)0, acc1, false, false);
    }

    const float bv0 = bias[n0 + lc];
    const float bv1 = bias[n0 + 16 + lc];
    float* c0 = C + (size_t)(m0 + hi * 8) * N + (n0 + lc);
    float* c1 = c0 + 16;
#pragma unroll
    for (int r = 0; r < 8; ++r) {
        c0[(size_t)r * N] = acc0[r] + bv0;
        c1[(size_t)r * N] = acc1[r] + bv1;
    }
}

// ---------------------------------------------------------------------------
// Persistent GRU recurrence for one layer.
//   gi   : [B*T, 3H] f32, row = b*T + t (b_ih already included)
//   Whh  : [3H, H] bf16 row-major   (B operand: column g of Whh^T = row g)
//   hcur : [2][B*H] f32  double-buffered state
//   hbuf : [2][B*H] bf16 double-buffered state (async-staged into LDS)
//   hout : [B*T, H] bf16 time series (relu-applied when relu_out)
// Grid = 16 workgroups (32 hidden cols each); device-wide barrier per step.
// Per step, the full 64x512 bf16 h matrix (64 KB) is DMA'd into LDS with
// GLOBAL_LOAD_ASYNC_TO_LDS_B128 (ASYNCcnt), XOR-swizzled by row so the
// 16-lane column-parallel ds_load_b128 A-fragment reads are bank-conflict
// free. Each wave owns one 16(batch) x 16(cols) tile of gh for ALL THREE
// gates, so the gate nonlinearity runs directly on the WMMA accumulators.
// ---------------------------------------------------------------------------
__global__ __launch_bounds__(256)
void k_gru_scan(const float* __restrict__ gi, const __bf16* __restrict__ Whh,
                const float* __restrict__ bhh,
                float* __restrict__ hcur, __bf16* __restrict__ hbuf,
                __bf16* __restrict__ hout, unsigned int* __restrict__ bar,
                int relu_out) {
    __shared__ __align__(16) __bf16 sh_h[BB * HH];   // 64 KB staged h

    const int tid  = threadIdx.x;
    const int lane = tid & 31;
    const int wave = tid >> 5;
    const int lc   = lane & 15;
    const int hi   = lane >> 4;
    const int m0   = (wave >> 1) * 16;                  // batch tile
    const int j0   = blockIdx.x * 32 + (wave & 1) * 16; // hidden col tile
    const int j    = j0 + lc;                           // this lane's column

    const __bf16* wr_r = Whh + (size_t)(j)          * HH + hi * 16;
    const __bf16* wr_z = Whh + (size_t)(HH  + j)    * HH + hi * 16;
    const __bf16* wr_n = Whh + (size_t)(2 * HH + j) * HH + hi * 16;
    const float bhr = bhh[j], bhz = bhh[HH + j], bhn = bhh[2 * HH + j];

    // LDS byte offset of sh_h: low 32 bits of the LDS-aperture address.
    const unsigned sh_base =
        (unsigned)(unsigned long long)(&sh_h[0]);

    const unsigned nblocks = gridDim.x;
    const int mA = m0 + lc;                             // A-frag row (= row&15 == lc)

    for (int t = 0; t < TT; ++t) {
        const int cur = t & 1, nxt = cur ^ 1;

        // --- Async-stage h[cur] (64x512 bf16) into LDS, swizzled ---------
        {
            const __bf16* hb = hbuf + (size_t)cur * (BB * HH);
#pragma unroll
            for (int it = 0; it < 16; ++it) {           // 4096 b128 xfers / 256 thr
                const int i     = tid + it * 256;
                const int row   = i >> 6;               // 64 chunks of 8 bf16 / row
                const int chunk = i & 63;
                const unsigned lds_addr =
                    sh_base + (unsigned)(row * (HH * 2) +
                                         ((chunk ^ (row & 15)) << 4));
                const unsigned long long g =
                    (unsigned long long)(hb + (size_t)row * HH + chunk * 8);
                asm volatile("global_load_async_to_lds_b128 %0, %1, off"
                             :: "v"(lds_addr), "v"(g) : "memory");
            }
            asm volatile("s_wait_asynccnt 0" ::: "memory");
            __syncthreads();
        }

        // --- gh = h @ Whh^T : A fragments from LDS, B from global (L2) ---
        v8f accR = {}, accZ = {}, accN = {};
        for (int k0 = 0; k0 < HH; k0 += 32) {
            const int c0 = (k0 >> 3) + 2 * hi;          // first 16B chunk index
            const __bf16* a0 = sh_h + (size_t)mA * HH + ((c0 ^ lc) << 3);
            const __bf16* a1 = sh_h + (size_t)mA * HH + (((c0 + 1) ^ lc) << 3);
            v16bf a, br, bz, bn;
            *(uint4*)&a        = *(const uint4*)a0;     // ds_load_b128
            *((uint4*)&a + 1)  = *(const uint4*)a1;     // ds_load_b128
            *(uint4*)&br       = *(const uint4*)(wr_r + k0);
            *((uint4*)&br + 1) = *(const uint4*)(wr_r + k0 + 8);
            *(uint4*)&bz       = *(const uint4*)(wr_z + k0);
            *((uint4*)&bz + 1) = *(const uint4*)(wr_z + k0 + 8);
            *(uint4*)&bn       = *(const uint4*)(wr_n + k0);
            *((uint4*)&bn + 1) = *(const uint4*)(wr_n + k0 + 8);
            accR = __builtin_amdgcn_wmma_f32_16x16x32_bf16(
                       false, a, false, br, (short)0, accR, false, false);
            accZ = __builtin_amdgcn_wmma_f32_16x16x32_bf16(
                       false, a, false, bz, (short)0, accZ, false, false);
            accN = __builtin_amdgcn_wmma_f32_16x16x32_bf16(
                       false, a, false, bn, (short)0, accN, false, false);
        }

        // --- Gate math on the C/D layout: vgpr r -> batch m0+hi*8+r, col j
#pragma unroll
        for (int r = 0; r < 8; ++r) {
            const int m = m0 + hi * 8 + r;
            const size_t grow = ((size_t)m * TT + t) * G3H;
            const float i_r = gi[grow + j];
            const float i_z = gi[grow + HH + j];
            const float i_n = gi[grow + 2 * HH + j];
            const float h_r = accR[r] + bhr;
            const float h_z = accZ[r] + bhz;
            const float h_n = accN[r] + bhn;
            const float rg = 1.0f / (1.0f + __expf(-(i_r + h_r)));
            const float zg = 1.0f / (1.0f + __expf(-(i_z + h_z)));
            const float ng = tanhf(i_n + rg * h_n);
            const float hp = hcur[(size_t)cur * (BB * HH) + (size_t)m * HH + j];
            const float hn2 = (1.0f - zg) * ng + zg * hp;
            hcur[(size_t)nxt * (BB * HH) + (size_t)m * HH + j] = hn2;
            hbuf[(size_t)nxt * (BB * HH) + (size_t)m * HH + j] = (__bf16)hn2;
            const float ov = relu_out ? fmaxf(hn2, 0.0f) : hn2;
            hout[((size_t)m * TT + t) * HH + j] = (__bf16)ov;
        }

        // --- Device-wide barrier between time steps (16 resident WGs) ----
        __threadfence();
        __syncthreads();
        if (tid == 0) {
            __hip_atomic_fetch_add(bar, 1u, __ATOMIC_RELEASE,
                                   __HIP_MEMORY_SCOPE_AGENT);
            const unsigned target = (unsigned)(t + 1) * nblocks;
            while (__hip_atomic_load(bar, __ATOMIC_ACQUIRE,
                                     __HIP_MEMORY_SCOPE_AGENT) < target) {
                __builtin_amdgcn_s_sleep(2);
            }
        }
        __syncthreads();
    }
}

// ---------------------------------------------------------------------------
// Host launcher
// ---------------------------------------------------------------------------
extern "C" void kernel_launch(void* const* d_in, const int* in_sizes, int n_in,
                              void* d_out, int out_size, void* d_ws, size_t ws_size,
                              hipStream_t stream) {
    (void)in_sizes; (void)n_in; (void)out_size; (void)ws_size;

    const float* x     = (const float*)d_in[0];
    const float* tin   = (const float*)d_in[1];
    const float* Wx    = (const float*)d_in[2];
    const float* bx    = (const float*)d_in[3];
    const float* Wt    = (const float*)d_in[4];
    const float* bt    = (const float*)d_in[5];
    const float* W_ih0 = (const float*)d_in[6];
    const float* W_hh0 = (const float*)d_in[7];
    const float* b_ih0 = (const float*)d_in[8];
    const float* b_hh0 = (const float*)d_in[9];
    const float* W_ih1 = (const float*)d_in[10];
    const float* W_hh1 = (const float*)d_in[11];
    const float* b_ih1 = (const float*)d_in[12];
    const float* b_hh1 = (const float*)d_in[13];
    const float* Wo    = (const float*)d_in[14];
    const float* bo    = (const float*)d_in[15];
    float* out = (float*)d_out;

    // Workspace carve-out (256B aligned slices)
    char* ws = (char*)d_ws;
    auto carve = [&](size_t bytes) -> void* {
        void* p = (void*)ws;
        ws += (bytes + 255) & ~(size_t)255;
        return p;
    };
    __bf16* Wih0b = (__bf16*)carve((size_t)G3H * D_MODEL * 2);
    __bf16* Whh0b = (__bf16*)carve((size_t)G3H * HH * 2);
    __bf16* Wih1b = (__bf16*)carve((size_t)G3H * HH * 2);
    __bf16* Whh1b = (__bf16*)carve((size_t)G3H * HH * 2);
    __bf16* Wob   = (__bf16*)carve((size_t)D_OUT * HH * 2);
    __bf16* xpb   = (__bf16*)carve((size_t)ROWS * D_MODEL * 2);
    __bf16* h1b   = (__bf16*)carve((size_t)ROWS * HH * 2);   // reused as relu(h2)
    float*  gi    = (float*)carve((size_t)ROWS * G3H * 4);   // reused for gi0, gi1
    // Recurrence state block (zeroed before each scan)
    float*        hcur = (float*)carve((size_t)2 * BB * HH * 4);
    __bf16*       hbuf = (__bf16*)carve((size_t)2 * BB * HH * 2);
    unsigned int* bar  = (unsigned int*)carve(256);
    const int state_u32 = (2 * BB * HH) + (2 * BB * HH / 2) + 64; // hcur+hbuf+bar

    // 1) Weights -> bf16
    auto cvt = [&](const float* s, __bf16* d, int n) {
        k_f32_to_bf16<<<(n + 255) / 256, 256, 0, stream>>>(s, d, n);
    };
    cvt(W_ih0, Wih0b, G3H * D_MODEL);
    cvt(W_hh0, Whh0b, G3H * HH);
    cvt(W_ih1, Wih1b, G3H * HH);
    cvt(W_hh1, Whh1b, G3H * HH);
    cvt(Wo,    Wob,   D_OUT * HH);

    // 2) Input projection -> xp (bf16)
    k_input_proj<<<(ROWS * D_MODEL) / 256, 256, 0, stream>>>(
        x, tin, Wx, bx, Wt, bt, xpb);

    // 3) gi0 = xp @ W_ih0^T + b_ih0        (32768 x 1536, K=256)
    k_gemm_bf16<<<dim3(ROWS / 64, G3H / 64), 256, 0, stream>>>(
        xpb, Wih0b, b_ih0, gi, D_MODEL, G3H);

    // 4) GRU layer 0 scan -> h1 (bf16 series)
    k_zero_u32<<<(state_u32 + 255) / 256, 256, 0, stream>>>(
        (unsigned int*)hcur, state_u32);
    k_gru_scan<<<HH / 32, 256, 0, stream>>>(
        gi, Whh0b, b_hh0, hcur, hbuf, h1b, bar, /*relu=*/0);

    // 5) gi1 = h1 @ W_ih1^T + b_ih1        (32768 x 1536, K=512)
    k_gemm_bf16<<<dim3(ROWS / 64, G3H / 64), 256, 0, stream>>>(
        h1b, Wih1b, b_ih1, gi, HH, G3H);

    // 6) GRU layer 1 scan -> relu(h2) (bf16 series, overwrites h1b)
    k_zero_u32<<<(state_u32 + 255) / 256, 256, 0, stream>>>(
        (unsigned int*)hcur, state_u32);
    k_gru_scan<<<HH / 32, 256, 0, stream>>>(
        gi, Whh1b, b_hh1, hcur, hbuf, h1b, bar, /*relu=*/1);

    // 7) out = relu(h2) @ Wo^T + bo        (32768 x 64, K=512)
    k_gemm_bf16<<<dim3(ROWS / 64, D_OUT / 64), 256, 0, stream>>>(
        h1b, Wob, bo, out, HH, D_OUT);
}
